// Decoder_79508434583915
// MI455X (gfx1250) — matrix operand order repaired and checked by reading, not compiled, not run on hardware
//
#include <hip/hip_runtime.h>
#include <math.h>

// Problem constants (from reference): B=64, S=1024, E=128, H=8, DK=16
#define B_   64
#define S_   1024
#define E_   128
#define H_   8
#define DK_  16
#define P3E  384   // 3*E

typedef __attribute__((ext_vector_type(16))) __bf16 v16bf;
typedef __attribute__((ext_vector_type(8)))  float  v8f;

// ---------- bf16 helpers ----------
static __device__ __forceinline__ unsigned short f2bf(float f) {
    unsigned u = __float_as_uint(f);
    unsigned r = 0x7fffu + ((u >> 16) & 1u);           // round-to-nearest-even
    return (unsigned short)((u + r) >> 16);
}
static __device__ __forceinline__ float bflo(unsigned u) { return __uint_as_float(u << 16); }
static __device__ __forceinline__ float bfhi(unsigned u) { return __uint_as_float(u & 0xffff0000u); }
static __device__ __forceinline__ __bf16 us2bf(unsigned short u) {
    union { unsigned short s; __bf16 b; } c; c.s = u; return c.b;
}

// ---------- 1) convert encoded -> bf16, W_node -> bf16 transposed [384,128] ----------
__global__ void prep_kernel(const float* __restrict__ enc, const float* __restrict__ Wnode,
                            unsigned short* __restrict__ encb, unsigned short* __restrict__ WnT) {
    const long N1 = (long)B_ * S_ * E_;
    for (long i = blockIdx.x * (long)blockDim.x + threadIdx.x; i < N1;
         i += (long)gridDim.x * blockDim.x)
        encb[i] = f2bf(enc[i]);
    const int N2 = P3E * E_;
    for (int i = blockIdx.x * blockDim.x + threadIdx.x; i < N2; i += gridDim.x * blockDim.x) {
        int n = i / E_, k = i % E_;
        WnT[i] = f2bf(Wnode[k * P3E + n]);
    }
}

// ---------- 2) h_hat = mean_s(encoded) @ W_ctx ----------
__global__ void hhat_kernel(const float* __restrict__ enc, const float* __restrict__ Wctx,
                            float* __restrict__ hhat) {
    __shared__ float mean[E_];
    const int b = blockIdx.x, e = threadIdx.x;
    float s = 0.f;
    const float* p = enc + (size_t)b * S_ * E_ + e;
    for (int i = 0; i < S_; ++i) s += p[(size_t)i * E_];
    mean[e] = s * (1.f / (float)S_);
    __syncthreads();
    float acc = 0.f;
    #pragma unroll 4
    for (int k = 0; k < E_; ++k) acc += mean[k] * Wctx[k * E_ + e];
    hhat[b * E_ + e] = acc;
}

// ---------- 3) proj = encoded @ W_node via v_wmma_f32_16x16x32_bf16 ----------
// A: [M=65536,128] bf16 row-major. BT: [384,128] bf16 (N-major). C: [M,384] bf16.
// Per-lane fragment layouts follow CDNA5 ISA 7.12.2:
//  A 16x32 bf16: lane<16 row=lane holds K 0-7 (halfs 0-7) and K 16-23 (halfs 8-15);
//                lane>=16 row=lane-16 holds K 8-15 and K 24-31.
//  B 32x16 bf16: lane<16 N=lane holds K 0-7 and K 16-23; lane>=16 N=lane-16 holds K 8-15 and K 24-31.
//  C 16x16 f32:  VGPR r, lane -> M = r + (lane>=16 ? 8 : 0), N = lane&15.
__global__ void __launch_bounds__(256)
proj_gemm_kernel(const unsigned short* __restrict__ A,
                 const unsigned short* __restrict__ BT,
                 unsigned short* __restrict__ C) {
    const int M  = B_ * S_;
    const int NT = P3E / 16;                 // 24 tiles along N
    const int wave = threadIdx.x >> 5;
    const int lane = threadIdx.x & 31;
    const int tile = blockIdx.x * 8 + wave;  // one 16x16 tile per wave
    const int tm = tile / NT, tn = tile % NT;
    if (tm >= M / 16) return;
    const int m0 = tm * 16, n0 = tn * 16;
    const int row  = lane & 15;
    const int hi16 = (lane >> 4) & 1;

    v8f acc = {};
    const unsigned short* arow = A  + (size_t)(m0 + row) * E_ + (hi16 ? 8 : 0);
    const unsigned short* bcol = BT + (size_t)(n0 + row) * E_ + (hi16 ? 8 : 0);

    #pragma unroll
    for (int kc = 0; kc < E_; kc += 32) {
        v16bf a, bm;
        #pragma unroll
        for (int i = 0; i < 8; ++i) {
            a[i]     = us2bf(arow[kc + i]);
            a[i + 8] = us2bf(arow[kc + 16 + i]);
        }
        #pragma unroll
        for (int i = 0; i < 8; ++i) { bm[i] = us2bf(bcol[kc + i]); bm[i + 8] = us2bf(bcol[kc + 16 + i]); }
        acc = __builtin_amdgcn_wmma_f32_16x16x32_bf16(false, a, false, bm,
                                                      (short)0, acc, false, false);
    }
    unsigned short* crow = C + (size_t)(m0 + (hi16 ? 8 : 0)) * P3E + n0 + row;
    #pragma unroll
    for (int r = 0; r < 8; ++r) crow[(size_t)r * P3E] = f2bf(acc[r]);
}

// ---------- 4) persistent greedy decode: 64 blocks (one per batch), 1024 steps ----------
__global__ void __launch_bounds__(256)
decode_kernel(const float* __restrict__ enc,        // [B,S,E] f32
              const float* __restrict__ Wupd,       // [256,128]
              const float* __restrict__ Wout,       // [128,128]
              const float* __restrict__ Wph,        // [256]
              const float* __restrict__ hhat,       // [B,128]
              const unsigned short* __restrict__ proj, // [B,S,384] bf16 (gK|gV|logitK)
              float* __restrict__ out) {            // [64] logp + [64*1024] solution
    __shared__ float sh_h[E_], sh_q[E_], sh_heads[E_], sh_g[E_], sh_emb[2 * E_];
    __shared__ float slog[S_];
    __shared__ unsigned smask[S_ / 32];
    __shared__ float rf[8]; __shared__ int ri[8]; __shared__ float rs[8];
    __shared__ float sMAX; __shared__ int sARG;
    __shared__ int sFirst, sLast; __shared__ float sLP;

    const int b = blockIdx.x;
    const int tid = threadIdx.x;
    const int wave = tid >> 5, lane = tid & 31;

    if (tid < S_ / 32) smask[tid] = 0u;
    if (tid < E_) sh_h[tid] = hhat[b * E_ + tid];
    if (tid == 0) { sFirst = 0; sLast = 0; sLP = 0.f; }
    __syncthreads();

    const unsigned short* projb = proj + (size_t)b * S_ * P3E;
    const float inv_dk = 0.25f;          // 1/sqrt(16)
    const float inv_e  = 0.0883883476f;  // 1/sqrt(128)

    for (int t = 0; t < S_; ++t) {
        // ---- first/last embedding (placeholder at t==0) ----
        if (t == 0) {
            if (tid < 2 * E_) sh_emb[tid] = Wph[tid];
        } else {
            if (tid < E_) {
                sh_emb[tid]      = enc[((size_t)b * S_ + sFirst) * E_ + tid];
                sh_emb[E_ + tid] = enc[((size_t)b * S_ + sLast)  * E_ + tid];
            }
        }
        __syncthreads();

        // ---- q = h_hat + embfl @ W_upd ----
        if (tid < E_) {
            float q = sh_h[tid];
            #pragma unroll 4
            for (int k = 0; k < 2 * E_; ++k) q += sh_emb[k] * Wupd[k * E_ + tid];
            sh_q[tid] = q;
        }
        __syncthreads();

        // ---- attention: wave h handles head h; online softmax, single pass ----
        {
            const int h = wave;
            float qh[DK_];
            #pragma unroll
            for (int d = 0; d < DK_; ++d) qh[d] = sh_q[h * DK_ + d];
            float m = -INFINITY, sum = 0.f;
            float acc[DK_];
            #pragma unroll
            for (int d = 0; d < DK_; ++d) acc[d] = 0.f;

            for (int c = 0; c < S_ / 32; ++c) {
                const int s = c * 32 + lane;
                const unsigned mb = (smask[c] >> lane) & 1u;
                const unsigned short* pk = projb + (size_t)s * P3E + h * DK_;
                const uint4 k0 = *(const uint4*)pk;
                const uint4 k1 = *(const uint4*)(pk + 8);
                float v =
                    qh[0]*bflo(k0.x) + qh[1]*bfhi(k0.x) + qh[2]*bflo(k0.y) + qh[3]*bfhi(k0.y) +
                    qh[4]*bflo(k0.z) + qh[5]*bfhi(k0.z) + qh[6]*bflo(k0.w) + qh[7]*bfhi(k0.w) +
                    qh[8]*bflo(k1.x) + qh[9]*bfhi(k1.x) + qh[10]*bflo(k1.y) + qh[11]*bfhi(k1.y) +
                    qh[12]*bflo(k1.z) + qh[13]*bfhi(k1.z) + qh[14]*bflo(k1.w) + qh[15]*bfhi(k1.w);
                v *= inv_dk;
                if (!mb) {
                    const float nm = fmaxf(m, v);
                    const float sc = __expf(m - nm);   // 0 when m==-inf
                    const float e  = __expf(v - nm);
                    sum = sum * sc + e;
                    const unsigned short* pv = projb + (size_t)s * P3E + E_ + h * DK_;
                    const uint4 v0 = *(const uint4*)pv;
                    const uint4 v1 = *(const uint4*)(pv + 8);
                    const float vv[DK_] = {
                        bflo(v0.x), bfhi(v0.x), bflo(v0.y), bfhi(v0.y),
                        bflo(v0.z), bfhi(v0.z), bflo(v0.w), bfhi(v0.w),
                        bflo(v1.x), bfhi(v1.x), bflo(v1.y), bfhi(v1.y),
                        bflo(v1.z), bfhi(v1.z), bflo(v1.w), bfhi(v1.w) };
                    #pragma unroll
                    for (int d = 0; d < DK_; ++d) acc[d] = acc[d] * sc + e * vv[d];
                    m = nm;
                }
            }
            // butterfly merge of online-softmax states across the wave32
            #pragma unroll
            for (int off = 16; off >= 1; off >>= 1) {
                const float om = __shfl_xor(m, off, 32);
                const float os = __shfl_xor(sum, off, 32);
                float oa[DK_];
                #pragma unroll
                for (int d = 0; d < DK_; ++d) oa[d] = __shfl_xor(acc[d], off, 32);
                const float M2 = fmaxf(m, om);
                const float w1 = (m  > -INFINITY) ? __expf(m  - M2) : 0.f;
                const float w2 = (om > -INFINITY) ? __expf(om - M2) : 0.f;
                sum = sum * w1 + os * w2;
                #pragma unroll
                for (int d = 0; d < DK_; ++d) acc[d] = acc[d] * w1 + oa[d] * w2;
                m = M2;
            }
            if (lane == 0) {
                const float inv = 1.f / sum;
                #pragma unroll
                for (int d = 0; d < DK_; ++d) sh_heads[h * DK_ + d] = acc[d] * inv;
            }
        }
        __syncthreads();

        // ---- glimpse = heads @ W_out ----
        if (tid < E_) {
            float g = 0.f;
            #pragma unroll 4
            for (int k = 0; k < E_; ++k) g += sh_heads[k] * Wout[k * E_ + tid];
            sh_g[tid] = g;
        }
        __syncthreads();

        // ---- logits + argmax (first-index tie-break, like jnp.argmax) ----
        float lmax = -INFINITY; int larg = 0x7fffffff;
        #pragma unroll
        for (int j = 0; j < 4; ++j) {
            const int s = tid + j * 256;
            const unsigned short* pl = projb + (size_t)s * P3E + 2 * E_;
            float d = 0.f;
            #pragma unroll 4
            for (int i = 0; i < E_ / 8; ++i) {
                const uint4 w = *(const uint4*)(pl + i * 8);
                d += sh_g[i*8+0]*bflo(w.x) + sh_g[i*8+1]*bfhi(w.x)
                   + sh_g[i*8+2]*bflo(w.y) + sh_g[i*8+3]*bfhi(w.y)
                   + sh_g[i*8+4]*bflo(w.z) + sh_g[i*8+5]*bfhi(w.z)
                   + sh_g[i*8+6]*bflo(w.w) + sh_g[i*8+7]*bfhi(w.w);
            }
            float lv = tanhf(d * inv_e) * 10.0f;
            if ((smask[s >> 5] >> (s & 31)) & 1u) lv = -INFINITY;
            slog[s] = lv;
            if (lv > lmax || (lv == lmax && s < larg)) { lmax = lv; larg = s; }
        }
        #pragma unroll
        for (int off = 16; off >= 1; off >>= 1) {
            const float ov = __shfl_xor(lmax, off, 32);
            const int   oi = __shfl_xor(larg, off, 32);
            if (ov > lmax || (ov == lmax && oi < larg)) { lmax = ov; larg = oi; }
        }
        if (lane == 0) { rf[wave] = lmax; ri[wave] = larg; }
        __syncthreads();
        if (tid == 0) {
            float mv = rf[0]; int mi = ri[0];
            for (int w = 1; w < 8; ++w)
                if (rf[w] > mv || (rf[w] == mv && ri[w] < mi)) { mv = rf[w]; mi = ri[w]; }
            sMAX = mv; sARG = mi;
        }
        __syncthreads();

        // ---- sum(exp(logits - max)), exact masked log-softmax of chosen city ----
        float mysum = 0.f;
        #pragma unroll
        for (int j = 0; j < 4; ++j) mysum += __expf(slog[tid + j * 256] - sMAX);
        #pragma unroll
        for (int off = 16; off >= 1; off >>= 1) mysum += __shfl_xor(mysum, off, 32);
        if (lane == 0) rs[wave] = mysum;
        __syncthreads();
        if (tid == 0) {
            float tot = 0.f;
            for (int w = 0; w < 8; ++w) tot += rs[w];
            const int city = sARG;
            sLP += slog[city] - sMAX - logf(tot);
            smask[city >> 5] |= (1u << (city & 31));
            sLast = city;
            if (t == 0) sFirst = city;
            out[64 + (size_t)b * S_ + t] = (float)city;   // solution
        }
        __syncthreads();
    }
    if (tid == 0) out[b] = sLP;                            // log_probabilities
}

// ---------- host launch ----------
extern "C" void kernel_launch(void* const* d_in, const int* in_sizes, int n_in,
                              void* d_out, int out_size, void* d_ws, size_t ws_size,
                              hipStream_t stream) {
    (void)in_sizes; (void)n_in; (void)out_size; (void)ws_size;
    const float* enc   = (const float*)d_in[0];
    const float* Wctx  = (const float*)d_in[1];
    const float* Wupd  = (const float*)d_in[2];
    const float* Wnode = (const float*)d_in[3];
    const float* Wout  = (const float*)d_in[4];
    const float* Wph   = (const float*)d_in[5];
    float* out = (float*)d_out;

    char* ws = (char*)d_ws;
    size_t off = 0;
    auto alloc = [&](size_t bytes) -> void* {
        void* p = ws + off;
        off = (off + bytes + 255) & ~(size_t)255;
        return p;
    };
    float*          hhat = (float*)         alloc((size_t)B_ * E_ * sizeof(float));
    unsigned short* encb = (unsigned short*)alloc((size_t)B_ * S_ * E_ * 2);
    unsigned short* WnT  = (unsigned short*)alloc((size_t)P3E * E_ * 2);
    unsigned short* proj = (unsigned short*)alloc((size_t)B_ * S_ * P3E * 2);

    prep_kernel<<<4096, 256, 0, stream>>>(enc, Wnode, encb, WnT);
    hhat_kernel<<<B_, E_, 0, stream>>>(enc, Wctx, hhat);
    const int tiles = (B_ * S_ / 16) * (P3E / 16);     // 98304 (exactly divisible by 8)
    proj_gemm_kernel<<<tiles / 8, 256, 0, stream>>>(encb, WnT, proj);
    decode_kernel<<<B_, 256, 0, stream>>>(enc, Wupd, Wout, Wph, hhat, proj, out);
}
